// TemporalHealthComplex_68599217652417
// MI455X (gfx1250) — matrix operand order repaired and checked by reading, not compiled, not run on hardware
//
#include <hip/hip_runtime.h>
#include <hip/hip_bf16.h>
#include <math.h>

// ---------------------------------------------------------------------------
// Types for CDNA5 WMMA (wave32): D(16x16,f32) = A(16x32,bf16) x B(32x16,bf16) + C
// ---------------------------------------------------------------------------
typedef __attribute__((ext_vector_type(16))) __bf16 v16bf;
typedef __attribute__((ext_vector_type(8)))  float  v8f;

#define BM 128
#define BN 128
#define BK 32
#define LDSS 40   // padded LDS row stride (elements) to spread banks

__device__ __forceinline__ unsigned short f2bf(float f) {
  union { float f; unsigned u; } x; x.f = f;
  unsigned u = x.u;
  return (unsigned short)((u + 0x7FFFu + ((u >> 16) & 1u)) >> 16); // RNE
}

// ---------------------------------------------------------------------------
// Generic WMMA GEMM:  val = [acc? Cf] + [bias] + sign * sum_taps A(shifted) W_tap^T
//   A:  bf16 [M x lda] (row-major), row shift per tap with per-batch causal mask
//   W:  bf16 [ntaps][N][K] (row-major, K contiguous)
//   op: 0 = none, 1 = exact GELU, 2 = sigmoid,
//       3 = final: out = aux1 + sigmoid(val) * aux2
//   Cf (f32) and/or Cb (bf16) written if non-null.
// ---------------------------------------------------------------------------
__global__ __launch_bounds__(256)
void gemm_wmma_bf16(const unsigned short* __restrict__ A, int lda,
                    const unsigned short* __restrict__ W,
                    const float* __restrict__ bias,
                    float* __restrict__ Cf,
                    unsigned short* __restrict__ Cb,
                    int ldc,
                    const float* __restrict__ aux1,
                    const float* __restrict__ aux2,
                    int M, int N, int K,
                    int ntaps, int shift0, int batchS,
                    float sign, int accFlag, int op)
{
  __shared__ unsigned short As[BM * LDSS];
  __shared__ unsigned short Ws[BN * LDSS];

  const int tid   = threadIdx.x;
  const int lane  = tid & 31;
  const int wave  = tid >> 5;
  const int wm    = wave & 3;    // 4 wave-rows
  const int wn    = wave >> 2;   // 2 wave-cols
  const int mblk  = blockIdx.y * BM;
  const int nblk  = blockIdx.x * BN;
  const int halfk = lane >> 4;   // 0/1
  const int l15   = lane & 15;

  const int lr = tid >> 1;           // 0..127: tile row loaded by this thread
  const int lc = (tid & 1) * 16;     // 0 / 16 element offset

  v8f acc[2][4];
  #pragma unroll
  for (int s = 0; s < 2; ++s)
    #pragma unroll
    for (int j = 0; j < 4; ++j)
      #pragma unroll
      for (int v = 0; v < 8; ++v) acc[s][j][v] = 0.0f;

  for (int tap = 0; tap < ntaps; ++tap) {
    const unsigned short* Wt = W + (size_t)tap * N * K;
    const int shift = shift0 + tap;                 // <= 0 (causal)
    const int grow  = mblk + lr;                    // output row this thread loads
    const bool valid = ((grow % batchS) + shift) >= 0;
    const unsigned short* Wrow = Wt + (size_t)(nblk + lr) * K;

    for (int k0 = 0; k0 < K; k0 += BK) {
      uint4 a0 = {0, 0, 0, 0}, a1 = {0, 0, 0, 0};
      if (valid) {
        const unsigned short* Arow = A + (size_t)(grow + shift) * lda;
        const uint4* p = (const uint4*)(Arow + k0 + lc);
        a0 = p[0]; a1 = p[1];
      }
      const uint4* q = (const uint4*)(Wrow + k0 + lc);
      uint4 w0 = q[0], w1 = q[1];

      __syncthreads();                              // prev-iter LDS reads done
      *(uint4*)&As[lr * LDSS + lc]     = a0;
      *(uint4*)&As[lr * LDSS + lc + 8] = a1;
      *(uint4*)&Ws[lr * LDSS + lc]     = w0;
      *(uint4*)&Ws[lr * LDSS + lc + 8] = w1;
      __syncthreads();                              // tiles visible

      union Frag { uint4 u[2]; v16bf v; };
      Frag af[2], bfr[4];
      // A frag: lane<16 holds K 0-7 & 16-23 of row (l15); lane>=16 holds K 8-15 & 24-31
      #pragma unroll
      for (int s = 0; s < 2; ++s) {
        const unsigned short* bp = &As[(wm * 32 + s * 16 + l15) * LDSS + halfk * 8];
        af[s].u[0] = *(const uint4*)(bp);
        af[s].u[1] = *(const uint4*)(bp + 16);
      }
      // B frag: lane<16 -> col n, K 0-7 & 16-23; lane>=16 -> K 8-15 & 24-31
      #pragma unroll
      for (int j = 0; j < 4; ++j) {
        const unsigned short* bp = &Ws[(wn * 64 + j * 16 + l15) * LDSS + halfk * 8];
        bfr[j].u[0] = *(const uint4*)(bp);
        bfr[j].u[1] = *(const uint4*)(bp + 16);
      }
      #pragma unroll
      for (int s = 0; s < 2; ++s)
        #pragma unroll
        for (int j = 0; j < 4; ++j)
          acc[s][j] = __builtin_amdgcn_wmma_f32_16x16x32_bf16(
              false, af[s].v, false, bfr[j].v, (short)0, acc[s][j], false, false);
    }
  }

  // Epilogue. D layout: lane l, VGPR v -> row (l>>4)*8 + v, col l&15 (per 16x16 tile)
  #pragma unroll
  for (int s = 0; s < 2; ++s) {
    #pragma unroll
    for (int j = 0; j < 4; ++j) {
      #pragma unroll
      for (int v = 0; v < 8; ++v) {
        const int gm = mblk + wm * 32 + s * 16 + halfk * 8 + v;
        const int gn = nblk + wn * 64 + j * 16 + l15;
        const size_t idx = (size_t)gm * ldc + gn;
        float val = acc[s][j][v] * sign;
        if (accFlag) val += Cf[idx];
        if (bias)    val += bias[gn];
        if (op == 1) {
          val = 0.5f * val * (1.0f + erff(val * 0.70710678118654752f));
        } else if (op == 2) {
          val = 1.0f / (1.0f + __expf(-val));
        } else if (op == 3) {
          const float g = 1.0f / (1.0f + __expf(-val));
          val = aux1[idx] + g * aux2[idx];
        }
        if (Cf) Cf[idx] = val;
        if (Cb) Cb[idx] = f2bf(val);
      }
    }
  }
}

// ---------------------------------------------------------------------------
// LayerNorm over E=512 per row; writes bf16 for the next GEMM's A operand.
// ---------------------------------------------------------------------------
__global__ __launch_bounds__(256)
void layernorm_to_bf16(const float* __restrict__ X, const float* __restrict__ g,
                       const float* __restrict__ b, unsigned short* __restrict__ out,
                       int E)
{
  __shared__ float s1[256], s2[256];
  const int row = blockIdx.x;
  const float* x = X + (size_t)row * E;
  const int t = threadIdx.x;
  const float a0 = x[t], a1 = x[t + 256];
  s1[t] = a0 + a1;
  s2[t] = a0 * a0 + a1 * a1;
  __syncthreads();
  for (int off = 128; off > 0; off >>= 1) {
    if (t < off) { s1[t] += s1[t + off]; s2[t] += s2[t + off]; }
    __syncthreads();
  }
  const float mu  = s1[0] / (float)E;
  const float var = s2[0] / (float)E - mu * mu;
  const float rs  = rsqrtf(var + 1e-5f);
  unsigned short* o = out + (size_t)row * E;
  o[t]       = f2bf((a0 - mu) * rs * g[t]       + b[t]);
  o[t + 256] = f2bf((a1 - mu) * rs * g[t + 256] + b[t + 256]);
}

__global__ void cvt_f32_bf16(const float* __restrict__ in,
                             unsigned short* __restrict__ out, size_t n)
{
  size_t i = (size_t)blockIdx.x * blockDim.x + threadIdx.x;
  const size_t stride = (size_t)gridDim.x * blockDim.x;
  for (; i < n; i += stride) out[i] = f2bf(in[i]);
}

// w[co][ci][k] f32  ->  wp[k][co][ci] bf16  (K=ci contiguous for B-fragments)
__global__ void pack_conv_w(const float* __restrict__ w,
                            unsigned short* __restrict__ wp, int C, int k)
{
  const int n = C * C * k;
  const int i = blockIdx.x * blockDim.x + threadIdx.x;
  if (i >= n) return;
  const int co  = i / (C * k);
  const int rem = i % (C * k);
  const int ci  = rem / k;
  const int j   = rem % k;
  wp[(size_t)j * C * C + (size_t)co * C + ci] = f2bf(w[i]);
}

__global__ void combine_bias(const float* __restrict__ a, const float* __restrict__ b,
                             float s, float* __restrict__ out, int n)
{
  const int i = blockIdx.x * blockDim.x + threadIdx.x;
  if (i < n) out[i] = a[i] + s * b[i];
}

// ---------------------------------------------------------------------------
extern "C" void kernel_launch(void* const* d_in, const int* in_sizes, int n_in,
                              void* d_out, int out_size, void* d_ws, size_t ws_size,
                              hipStream_t stream)
{
  (void)in_sizes; (void)n_in; (void)out_size; (void)ws_size;

  const float* x    = (const float*)d_in[0];
  const float* Wtc  = (const float*)d_in[1];
  const float* btc  = (const float*)d_in[2];
  const float* w1r  = (const float*)d_in[3];
  const float* b1r  = (const float*)d_in[4];
  const float* w1i  = (const float*)d_in[5];
  const float* b1i  = (const float*)d_in[6];
  const float* w2r  = (const float*)d_in[7];
  const float* b2r  = (const float*)d_in[8];
  const float* w2i  = (const float*)d_in[9];
  const float* b2i  = (const float*)d_in[10];
  const float* pw1  = (const float*)d_in[11];
  const float* pb1  = (const float*)d_in[12];
  const float* pw2  = (const float*)d_in[13];
  const float* pb2  = (const float*)d_in[14];
  const float* lng  = (const float*)d_in[15];
  const float* lnbt = (const float*)d_in[16];
  const float* Wout = (const float*)d_in[17];
  const float* bout = (const float*)d_in[18];
  const float* Wg   = (const float*)d_in[19];
  const float* bg   = (const float*)d_in[20];

  const int S = 4096, D = 2048, C = 256, E = 512, F = 1024;
  const int M = 4 * S;   // 16384 rows

  char* base = (char*)d_ws;
  size_t off = 0;
  auto alloc = [&](size_t bytes) -> void* {
    void* r = base + off;
    off = (off + bytes + 255) & ~(size_t)255;
    return r;
  };

  unsigned short* xb    = (unsigned short*)alloc((size_t)M * D * 2);
  unsigned short* Wtcb  = (unsigned short*)alloc((size_t)E * D * 2);
  unsigned short* Wgb   = (unsigned short*)alloc((size_t)D * D * 2);
  unsigned short* Woutb = (unsigned short*)alloc((size_t)D * E * 2);
  unsigned short* pw1b  = (unsigned short*)alloc((size_t)F * E * 2);
  unsigned short* pw2b  = (unsigned short*)alloc((size_t)E * F * 2);
  unsigned short* w1rp  = (unsigned short*)alloc((size_t)3 * C * C * 2);
  unsigned short* w1ip  = (unsigned short*)alloc((size_t)5 * C * C * 2);
  unsigned short* w2rp  = (unsigned short*)alloc((size_t)5 * C * C * 2);
  unsigned short* w2ip  = (unsigned short*)alloc((size_t)9 * C * C * 2);
  float*          cf    = (float*)         alloc((size_t)M * E * 4);
  unsigned short* cfb   = (unsigned short*)alloc((size_t)M * E * 2);
  float*          y1    = (float*)         alloc((size_t)M * E * 4);
  unsigned short* y1b   = (unsigned short*)alloc((size_t)M * E * 2);
  float*          co    = (float*)         alloc((size_t)M * E * 4);
  unsigned short* cob   = (unsigned short*)alloc((size_t)M * E * 2);
  unsigned short* hb    = (unsigned short*)alloc((size_t)M * F * 2);
  unsigned short* lnb   = (unsigned short*)alloc((size_t)M * E * 2);
  float*          outp  = (float*)         alloc((size_t)M * D * 4);
  float*          bc1re = (float*)alloc(C * 4);
  float*          bc1im = (float*)alloc(C * 4);
  float*          bc2re = (float*)alloc(C * 4);
  float*          bc2im = (float*)alloc(C * 4);

  // --- operand conversion / packing ---
  cvt_f32_bf16<<<8192, 256, 0, stream>>>(x,    xb,    (size_t)M * D);
  cvt_f32_bf16<<<2048, 256, 0, stream>>>(Wtc,  Wtcb,  (size_t)E * D);
  cvt_f32_bf16<<<4096, 256, 0, stream>>>(Wg,   Wgb,   (size_t)D * D);
  cvt_f32_bf16<<<2048, 256, 0, stream>>>(Wout, Woutb, (size_t)D * E);
  cvt_f32_bf16<<<1024, 256, 0, stream>>>(pw1,  pw1b,  (size_t)F * E);
  cvt_f32_bf16<<<1024, 256, 0, stream>>>(pw2,  pw2b,  (size_t)E * F);
  pack_conv_w<<<(3 * C * C + 255) / 256, 256, 0, stream>>>(w1r, w1rp, C, 3);
  pack_conv_w<<<(5 * C * C + 255) / 256, 256, 0, stream>>>(w1i, w1ip, C, 5);
  pack_conv_w<<<(5 * C * C + 255) / 256, 256, 0, stream>>>(w2r, w2rp, C, 5);
  pack_conv_w<<<(9 * C * C + 255) / 256, 256, 0, stream>>>(w2i, w2ip, C, 9);
  combine_bias<<<1, 256, 0, stream>>>(b1r, b1i, -1.0f, bc1re, C);
  combine_bias<<<1, 256, 0, stream>>>(b1r, b1i, +1.0f, bc1im, C);
  combine_bias<<<1, 256, 0, stream>>>(b2r, b2i, -1.0f, bc2re, C);
  combine_bias<<<1, 256, 0, stream>>>(b2r, b2i, +1.0f, bc2im, C);

  auto gemm = [&](const unsigned short* A, int lda, const unsigned short* W,
                  const float* bias, float* Cf, unsigned short* Cb, int ldc,
                  const float* a1p, const float* a2p, int N, int K,
                  int ntaps, int shift0, float sign, int accf, int op) {
    dim3 grid(N / 128, M / 128);
    gemm_wmma_bf16<<<grid, 256, 0, stream>>>(A, lda, W, bias, Cf, Cb, ldc,
                                             a1p, a2p, M, N, K,
                                             ntaps, shift0, S, sign, accf, op);
  };

  // 1) token compress: cf = x @ Wtc^T + btc   (also bf16 mirror)
  gemm(xb, D, Wtcb, btc, cf, cfb, E, nullptr, nullptr, E, D, 1, 0, 1.0f, 0, 0);

  // 2) complex conv1 (xr = cfb[:,0:256], xi = cfb[:,256:512])
  gemm(cfb,       E, w1rp, bc1re,  y1,     nullptr, E, nullptr, nullptr, C, C, 3, -2, +1.0f, 0, 0);
  gemm(cfb + C,   E, w1ip, nullptr,y1,     y1b,     E, nullptr, nullptr, C, C, 5, -4, -1.0f, 1, 0);
  gemm(cfb + C,   E, w1rp, bc1im,  y1 + C, nullptr, E, nullptr, nullptr, C, C, 3, -2, +1.0f, 0, 0);
  gemm(cfb,       E, w1ip, nullptr,y1 + C, y1b + C, E, nullptr, nullptr, C, C, 5, -4, +1.0f, 1, 0);

  // 3) complex conv2
  gemm(y1b,       E, w2rp, bc2re,  co,     nullptr, E, nullptr, nullptr, C, C, 5, -4, +1.0f, 0, 0);
  gemm(y1b + C,   E, w2ip, nullptr,co,     cob,     E, nullptr, nullptr, C, C, 9, -8, -1.0f, 1, 0);
  gemm(y1b + C,   E, w2rp, bc2im,  co + C, nullptr, E, nullptr, nullptr, C, C, 5, -4, +1.0f, 0, 0);
  gemm(y1b,       E, w2ip, nullptr,co + C, cob + C, E, nullptr, nullptr, C, C, 9, -8, +1.0f, 1, 0);

  // 4) MLP: h = gelu(co @ pw1^T + pb1) [bf16 only], then co += h @ pw2^T + pb2
  gemm(cob, E, pw1b, pb1, nullptr, hb,      F, nullptr, nullptr, F, E, 1, 0, 1.0f, 0, 1);
  gemm(hb,  F, pw2b, pb2, co,      nullptr, E, nullptr, nullptr, E, F, 1, 0, 1.0f, 1, 0);

  // 5) LayerNorm -> bf16
  layernorm_to_bf16<<<M, 256, 0, stream>>>(co, lng, lnbt, lnb, E);

  // 6) out = ln @ Wout^T + bout
  gemm(lnb, E, Woutb, bout, outp, nullptr, D, nullptr, nullptr, D, E, 1, 0, 1.0f, 0, 0);

  // 7) final: d_out = x + sigmoid(x @ Wg^T + bg) * outp
  gemm(xb, D, Wgb, bg, (float*)d_out, nullptr, D, x, outp, D, D, 1, 0, 1.0f, 0, 3);
}
